// GNN_47107201302761
// MI455X (gfx1250) — matrix operand order repaired and checked by reading, not compiled, not run on hardware
//
#include <hip/hip_runtime.h>
#include <hip/hip_bf16.h>

#define NN 50000
#define NE 800000
#define IN_D 128
#define HID_D 256
#define OUT_D 64

typedef __attribute__((ext_vector_type(16))) __bf16 v16bf;
typedef __attribute__((ext_vector_type(8)))  float  v8f;

__device__ __forceinline__ unsigned short f2bf(float f) {
  unsigned int u = __float_as_uint(f);
  unsigned int r = (u + 0x7FFFu + ((u >> 16) & 1u)) >> 16;  // RNE
  return (unsigned short)r;
}

// ---------------- degree / normalization ----------------
__global__ void k_fill_f32(float* __restrict__ p, float v, int n) {
  int i = blockIdx.x * blockDim.x + threadIdx.x;
  if (i < n) p[i] = v;
}

__global__ void k_deg_edges(const int* __restrict__ dst, float* __restrict__ deg, int E) {
  int e = blockIdx.x * blockDim.x + threadIdx.x;
  if (e < E) unsafeAtomicAdd(&deg[dst[e]], 1.0f);
}

__global__ void k_rsqrt(const float* __restrict__ deg, float* __restrict__ dis, int n) {
  int i = blockIdx.x * blockDim.x + threadIdx.x;
  if (i < n) dis[i] = rsqrtf(deg[i]);   // deg >= 1 always (self loop)
}

// ---------------- dtype conversion ----------------
__global__ void k_f32_to_bf16(const float* __restrict__ in, unsigned short* __restrict__ out, int n) {
  int i = blockIdx.x * blockDim.x + threadIdx.x;
  if (i < n) out[i] = f2bf(in[i]);
}

// in: [K,N] row-major fp32  ->  out: [N,K] row-major bf16
__global__ void k_transpose_bf16(const float* __restrict__ in, unsigned short* __restrict__ out,
                                 int K, int N) {
  int i = blockIdx.x * blockDim.x + threadIdx.x;
  if (i >= K * N) return;
  int k = i / N, n = i % N;
  out[(size_t)n * K + k] = f2bf(in[i]);
}

// ---------------- WMMA GEMM: C[M,N] = A[M,K] * Bt[N,K]^T (bf16 in, f32 out) ----------------
__global__ void k_gemm_bf16_wmma(const unsigned short* __restrict__ A,
                                 const unsigned short* __restrict__ Bt,
                                 float* __restrict__ C,
                                 int M, int N, int K) {
  int wid    = blockIdx.x * (blockDim.x >> 5) + (threadIdx.x >> 5);
  int ntN    = N >> 4;
  int ntiles = (M >> 4) * ntN;
  if (wid >= ntiles) return;
  int tm   = wid / ntN;
  int tn   = wid % ntN;
  int lane = threadIdx.x & 31;
  int half = lane >> 4;   // 0: lanes 0-15, 1: lanes 16-31
  int lr   = lane & 15;

  const uint4* arow = reinterpret_cast<const uint4*>(A  + (size_t)(tm * 16 + lr) * K);
  const uint4* bcol = reinterpret_cast<const uint4*>(Bt + (size_t)(tn * 16 + lr) * K);

  union Frag { v16bf v; uint4 q[2]; };
  v8f acc = {};
  for (int k0 = 0; k0 < K; k0 += 32) {
    Frag a, b;
    // A 16x32 bf16 fragment: half0 holds K = k0+[0..7]+[16..23], half1 holds K = k0+[8..15]+[24..31]
    int aoff = k0 + half * 8;                 // bf16 element offset (uint4 = 8 elems)
    a.q[0] = arow[aoff >> 3];
    a.q[1] = arow[(aoff + 16) >> 3];
    // B 32x16 fragment via Bt[N,K]: half0 holds K = k0+[0..7]+[16..23], half1 holds K = k0+[8..15]+[24..31]
    int boff = k0 + half * 8;
    b.q[0] = bcol[boff >> 3];
    b.q[1] = bcol[(boff + 16) >> 3];
    acc = __builtin_amdgcn_wmma_f32_16x16x32_bf16(false, a.v, false, b.v,
                                                  (short)0, acc, false, false);
  }
  // C layout: lane lr -> column, VGPR r -> row (tm*16 + half*8 + r)
  float* c0 = C + (size_t)(tm * 16 + half * 8) * N + (tn * 16 + lr);
#pragma unroll
  for (int r = 0; r < 8; ++r) c0[(size_t)r * N] = acc[r];
}

// ---------------- aggregation ----------------
// agg[i,c] = dis[i]^2 * h[i,c] + bias[c]   (self-loop contribution + bias, full overwrite)
__global__ void k_selfloop_bias(const float* __restrict__ h, const float* __restrict__ dis,
                                const float* __restrict__ bias, float* __restrict__ agg,
                                long long total, int Dmask, int logD) {
  long long t = (long long)blockIdx.x * blockDim.x + threadIdx.x;
  if (t >= total) return;
  int i = (int)(t >> logD);
  int c = (int)(t & Dmask);
  float d = dis[i];
  agg[t] = d * d * h[t] + bias[c];
}

__global__ void k_agg_edges(const float* __restrict__ h, const int* __restrict__ src,
                            const int* __restrict__ dst, const float* __restrict__ dis,
                            float* __restrict__ agg, long long total, int Dmask, int logD) {
  long long t = (long long)blockIdx.x * blockDim.x + threadIdx.x;
  if (t >= total) return;
  int e = (int)(t >> logD);
  int c = (int)(t & Dmask);
  int s = src[e], d = dst[e];
  float norm = dis[s] * dis[d];
  unsafeAtomicAdd(&agg[((size_t)d << logD) + c], h[((size_t)s << logD) + c] * norm);
}

__global__ void k_relu_bf16(const float* __restrict__ in, unsigned short* __restrict__ out,
                            long long n) {
  long long i = (long long)blockIdx.x * blockDim.x + threadIdx.x;
  if (i < n) out[i] = f2bf(fmaxf(in[i], 0.0f));
}

// ---------------- log-softmax over 64 columns, one wave32 per row ----------------
__global__ void k_logsoftmax64(const float* __restrict__ in, float* __restrict__ out, int rows) {
  int wid  = threadIdx.x >> 5;
  int lane = threadIdx.x & 31;
  int row  = blockIdx.x * (blockDim.x >> 5) + wid;
  if (row >= rows) return;
  const float* r = in + (size_t)row * 64;
  float v0 = r[lane], v1 = r[lane + 32];
  float m = fmaxf(v0, v1);
#pragma unroll
  for (int off = 16; off >= 1; off >>= 1) m = fmaxf(m, __shfl_xor(m, off, 32));
  float s = __expf(v0 - m) + __expf(v1 - m);
#pragma unroll
  for (int off = 16; off >= 1; off >>= 1) s += __shfl_xor(s, off, 32);
  float ls = __logf(s);
  float* o = out + (size_t)row * 64;
  o[lane]      = v0 - m - ls;
  o[lane + 32] = v1 - m - ls;
}

extern "C" void kernel_launch(void* const* d_in, const int* in_sizes, int n_in,
                              void* d_out, int out_size, void* d_ws, size_t ws_size,
                              hipStream_t stream) {
  (void)in_sizes; (void)n_in; (void)out_size; (void)ws_size;
  const float* x  = (const float*)d_in[0];
  const int*   ei = (const int*)  d_in[1];
  const float* W1 = (const float*)d_in[2];
  const float* b1 = (const float*)d_in[3];
  const float* W2 = (const float*)d_in[4];
  const float* b2 = (const float*)d_in[5];
  const int* src = ei;
  const int* dst = ei + NE;
  float* out = (float*)d_out;

  char* ws = (char*)d_ws;
  size_t off = 0;
  auto alloc = [&](size_t bytes) -> void* {
    void* p = ws + off;
    off += (bytes + 255) & ~(size_t)255;
    return p;
  };
  float*          deg  = (float*)         alloc((size_t)NN * 4);
  float*          dis  = (float*)         alloc((size_t)NN * 4);
  unsigned short* w1t  = (unsigned short*)alloc((size_t)HID_D * IN_D * 2);   // [HID,IN]
  unsigned short* w2t  = (unsigned short*)alloc((size_t)OUT_D * HID_D * 2);  // [OUT,HID]
  unsigned short* xb   = (unsigned short*)alloc((size_t)NN * IN_D * 2);
  float*          hlin = (float*)         alloc((size_t)NN * HID_D * 4);
  float*          agg1 = (float*)         alloc((size_t)NN * HID_D * 4);
  // aliases over dead regions:
  unsigned short* h1b   = (unsigned short*)hlin;  // relu(h1) bf16, overlays dead hlin
  float*          h2lin = agg1;                    // GEMM2 out, overlays dead agg1
  float*          agg2  = (float*)xb;              // layer-2 agg, overlays dead xb

  const int T = 256;
  // normalization
  k_fill_f32 <<<(NN + T - 1) / T, T, 0, stream>>>(deg, 1.0f, NN);
  k_deg_edges<<<(NE + T - 1) / T, T, 0, stream>>>(dst, deg, NE);
  k_rsqrt    <<<(NN + T - 1) / T, T, 0, stream>>>(deg, dis, NN);
  // bf16 staging
  k_f32_to_bf16 <<<((NN * IN_D) + T - 1) / T, T, 0, stream>>>(x, xb, NN * IN_D);
  k_transpose_bf16<<<((IN_D * HID_D) + T - 1) / T, T, 0, stream>>>(W1, w1t, IN_D, HID_D);
  k_transpose_bf16<<<((HID_D * OUT_D) + T - 1) / T, T, 0, stream>>>(W2, w2t, HID_D, OUT_D);
  // layer 1: GEMM -> self-loop+bias -> edge scatter -> relu
  {
    int tiles = (NN / 16) * (HID_D / 16);
    k_gemm_bf16_wmma<<<(tiles + 7) / 8, T, 0, stream>>>(xb, w1t, hlin, NN, HID_D, IN_D);
    long long tot = (long long)NN * HID_D;
    k_selfloop_bias<<<(int)((tot + T - 1) / T), T, 0, stream>>>(hlin, dis, b1, agg1, tot, HID_D - 1, 8);
    long long etot = (long long)NE * HID_D;
    k_agg_edges<<<(int)((etot + T - 1) / T), T, 0, stream>>>(hlin, src, dst, dis, agg1, etot, HID_D - 1, 8);
    k_relu_bf16<<<(int)((tot + T - 1) / T), T, 0, stream>>>(agg1, h1b, tot);
  }
  // layer 2: GEMM -> self-loop+bias -> edge scatter -> log_softmax
  {
    int tiles = (NN / 16) * (OUT_D / 16);
    k_gemm_bf16_wmma<<<(tiles + 7) / 8, T, 0, stream>>>(h1b, w2t, h2lin, NN, OUT_D, HID_D);
    long long tot = (long long)NN * OUT_D;
    k_selfloop_bias<<<(int)((tot + T - 1) / T), T, 0, stream>>>(h2lin, dis, b2, agg2, tot, OUT_D - 1, 6);
    long long etot = (long long)NE * OUT_D;
    k_agg_edges<<<(int)((etot + T - 1) / T), T, 0, stream>>>(h2lin, src, dst, dis, agg2, etot, OUT_D - 1, 6);
    k_logsoftmax64<<<(NN + 7) / 8, T, 0, stream>>>(agg2, out, NN);
  }
}
